// MultiHeadAttention_84877143703661
// MI455X (gfx1250) — matrix operand order repaired and checked by reading, not compiled, not run on hardware
//
#include <hip/hip_runtime.h>

typedef __attribute__((ext_vector_type(16))) _Float16 v16h;
typedef __attribute__((ext_vector_type(8)))  _Float16 v8h;
typedef __attribute__((ext_vector_type(8)))  float    v8f;
typedef __attribute__((ext_vector_type(4)))  float    v4f;

static constexpr int S  = 64;
static constexpr int D  = 128;
static constexpr int F  = 128;
static constexpr int H  = 8;
static constexpr int DK = 16;

// ---------------- Wk (f32 [F][D]) -> f16 ----------------
__global__ void k_cvt_wk(const float* __restrict__ Wk, _Float16* __restrict__ Wkh) {
    int i = blockIdx.x * 256 + threadIdx.x;
    if (i < F * D) Wkh[i] = (_Float16)Wk[i];
}

// ---------------- vbar[b][d] = mean_s v[b][s][d] ----------------
__global__ void k_vbar(const float* __restrict__ v, float* __restrict__ vbar, int B) {
    int i = blockIdx.x * 256 + threadIdx.x;   // i = b*D + d
    if (i >= B * D) return;
    int b = i >> 7, d = i & 127;
    const float* p = v + (size_t)b * S * D + d;
    float s = 0.f;
#pragma unroll 8
    for (int ss = 0; ss < S; ++ss) s += p[ss * D];
    vbar[i] = s * (1.0f / S);
}

// ---------------- q1[b][f] = vbar[b]·Wq[f] + bq[f] ----------------
__global__ void k_proj1(const float* __restrict__ vbar, const float* __restrict__ Wq,
                        const float* __restrict__ bq, float* __restrict__ q1) {
    __shared__ float sv[D];
    int b = blockIdx.x, f = threadIdx.x;
    sv[f] = vbar[b * D + f];
    __syncthreads();
    const float* w = Wq + f * D;
    float acc = bq[f];
#pragma unroll 8
    for (int d = 0; d < D; ++d) acc += sv[d] * w[d];
    q1[b * F + f] = acc;
}

// ---------------- BN batch stats -> scale/shift per feature ----------------
__global__ void k_bnstats(const float* __restrict__ q1, const float* __restrict__ gamma,
                          const float* __restrict__ beta, float* __restrict__ scale,
                          float* __restrict__ shift, int B) {
    __shared__ float s1[256];
    __shared__ float s2[256];
    int f = blockIdx.x, t = threadIdx.x;
    float a = 0.f, q = 0.f;
    for (int b = t; b < B; b += 256) {
        float x = q1[b * F + f];
        a += x; q += x * x;
    }
    s1[t] = a; s2[t] = q;
    __syncthreads();
    for (int off = 128; off > 0; off >>= 1) {
        if (t < off) { s1[t] += s1[t + off]; s2[t] += s2[t + off]; }
        __syncthreads();
    }
    if (t == 0) {
        float mu  = s1[0] / (float)B;
        float var = s2[0] / (float)B - mu * mu;   // biased, as BN uses
        float sc  = gamma[f] * rsqrtf(var + 1e-5f);
        scale[f] = sc;
        shift[f] = beta[f] - mu * sc;
    }
}

// ---------------- q2[b][g] = BN(q1[b])·W2[g] + b2[g] ----------------
__global__ void k_proj2(const float* __restrict__ q1, const float* __restrict__ scale,
                        const float* __restrict__ shift, const float* __restrict__ W2,
                        const float* __restrict__ b2, float* __restrict__ q2) {
    __shared__ float sq[F];
    int b = blockIdx.x, g = threadIdx.x;
    sq[g] = q1[b * F + g] * scale[g] + shift[g];
    __syncthreads();
    const float* w = W2 + g * F;
    float acc = b2[g];
#pragma unroll 8
    for (int f2 = 0; f2 < F; ++f2) acc += sq[f2] * w[f2];
    q2[b * F + g] = acc;
}

// ---------------- fused attention: one block per batch ----------------
// Per block: K = v[b]@Wk^T + bk (WMMA f16->f32), scores, softmax, out = attn@v
__global__ void __launch_bounds__(128) k_attn(
    const float* __restrict__ v, const _Float16* __restrict__ Wkh,
    const float* __restrict__ bk, const float* __restrict__ q2,
    float* __restrict__ outO, float* __restrict__ outA, int B) {

    __shared__ __align__(32) _Float16 sVh[S * D];   // 16 KB: v[b] in f16 (WMMA A)
    __shared__ __align__(16) float    sK[S * F];    // 32 KB: K matrix f32
    __shared__ float sQ[F];                         // q row for this batch
    __shared__ float sAttn[H * S];                  // logits -> probabilities

    const int b    = blockIdx.x;
    const int tid  = threadIdx.x;
    const int lane = tid & 31;
    const int wave = tid >> 5;     // 0..3 -> M-tile (16 seq rows each)
    const int lrow = lane & 15;    // m (A) / n (B) / col (C)
    const int lhi  = lane >> 4;    // k-half select

    // ---- stage v[b] into LDS as f16; stage q row ----
    const float* vb = v + (size_t)b * (S * D);
#pragma unroll 8
    for (int i = 0; i < (S * D) / 128; ++i) {
        int idx = i * 128 + tid;
        sVh[idx] = (_Float16)vb[idx];
    }
    sQ[tid] = q2[b * F + tid];
    __syncthreads();

    // ---- K = v @ Wk^T + bk via v_wmma_f32_16x16x32_f16 ----
    // A fragment (16x32 f16): lane holds row m=lrow of tile; halves 0-15 = k[16*lhi..16*lhi+15]
    // (two adjacent 16B chunks)
    v16h afrag[4];
#pragma unroll
    for (int kb = 0; kb < 4; ++kb) {
        const _Float16* abase = &sVh[(16 * wave + lrow) * D + 32 * kb + 16 * lhi];
        v8h lo = *(const v8h*)abase;
        v8h hi = *(const v8h*)(abase + 8);
        afrag[kb] = __builtin_shufflevector(lo, hi,
            0, 1, 2, 3, 4, 5, 6, 7, 8, 9, 10, 11, 12, 13, 14, 15);
    }
#pragma unroll
    for (int nt = 0; nt < 8; ++nt) {
        float bias = bk[16 * nt + lrow];          // bias depends on output feature only
        v8f acc;
#pragma unroll
        for (int r = 0; r < 8; ++r) acc[r] = bias;
#pragma unroll
        for (int kb = 0; kb < 4; ++kb) {
            // B fragment (32x16 f16): B[d][f] = Wk[f][d]; lane = col n + 16*k-half;
            // 16 contiguous halves along d of Wk row -> single 32B load, no transpose.
            const _Float16* bbase = &Wkh[(16 * nt + lrow) * D + 32 * kb + 16 * lhi];
            v16h bfrag = *(const v16h*)bbase;
            acc = __builtin_amdgcn_wmma_f32_16x16x32_f16(
                false, afrag[kb], false, bfrag, (short)0, acc, false, false);
        }
        // C/D layout: row = r + 8*lhi, col = lrow
#pragma unroll
        for (int r = 0; r < 8; ++r)
            sK[(16 * wave + 8 * lhi + r) * F + 16 * nt + lrow] = acc[r];
    }
    __syncthreads();

    // ---- scores: logit[h][s] = (q[h]·k[s,h]) / sqrt(16) ----
#pragma unroll
    for (int i = 0; i < 4; ++i) {
        int idx = i * 128 + tid;            // idx = h*64 + s
        int h = idx >> 6, s = idx & 63;
        const float* kr = &sK[s * F + h * DK];
        const float* qr = &sQ[h * DK];
        float acc = 0.f;
#pragma unroll
        for (int j = 0; j < DK; ++j) acc += qr[j] * kr[j];
        sAttn[idx] = acc * 0.25f;
    }
    __syncthreads();

    // ---- softmax over s: 16 threads per head, shfl reductions within width 16 ----
    {
        int h = tid >> 4, si = tid & 15;
        float x[4];
        float m = -1e30f;
#pragma unroll
        for (int i = 0; i < 4; ++i) { x[i] = sAttn[h * S + si + 16 * i]; m = fmaxf(m, x[i]); }
#pragma unroll
        for (int off = 8; off >= 1; off >>= 1) m = fmaxf(m, __shfl_xor(m, off, 16));
        float sum = 0.f;
#pragma unroll
        for (int i = 0; i < 4; ++i) { x[i] = __expf(x[i] - m); sum += x[i]; }
#pragma unroll
        for (int off = 8; off >= 1; off >>= 1) sum += __shfl_xor(sum, off, 16);
        float inv = 1.0f / sum;
#pragma unroll
        for (int i = 0; i < 4; ++i) {
            float a = x[i] * inv;
            sAttn[h * S + si + 16 * i] = a;
            outA[((size_t)h * B + b) * S + si + 16 * i] = a;   // attn output [H,B,S]
        }
    }
    __syncthreads();

    // ---- out[h][d] = sum_s attn[h][s] * v[b][s][d]  (f32, v from global/L2) ----
    {
        int h = tid >> 4, d0 = (tid & 15) * 8;
        v4f acc0 = {0.f, 0.f, 0.f, 0.f};
        v4f acc1 = {0.f, 0.f, 0.f, 0.f};
        const float* vp = vb + d0;
#pragma unroll 4
        for (int s = 0; s < S; ++s) {
            float a = sAttn[h * S + s];
            v4f x0 = *(const v4f*)(vp + s * D);
            v4f x1 = *(const v4f*)(vp + s * D + 4);
            acc0 += x0 * a;
            acc1 += x1 * a;
        }
        float* op = outO + ((size_t)h * B + b) * D + d0;       // out [H,B,D]
        *(v4f*)op       = acc0;
        *(v4f*)(op + 4) = acc1;
    }
}

extern "C" void kernel_launch(void* const* d_in, const int* in_sizes, int n_in,
                              void* d_out, int out_size, void* d_ws, size_t ws_size,
                              hipStream_t stream) {
    const float* v     = (const float*)d_in[0];
    const float* Wq    = (const float*)d_in[1];
    const float* bq    = (const float*)d_in[2];
    const float* gamma = (const float*)d_in[3];
    const float* beta  = (const float*)d_in[4];
    const float* W2    = (const float*)d_in[5];
    const float* b2    = (const float*)d_in[6];
    const float* Wk    = (const float*)d_in[7];
    const float* bk    = (const float*)d_in[8];
    const int B = in_sizes[0] / (S * D);   // 2048

    // workspace layout (floats); d_ws is 256B aligned from hipMalloc
    float* ws    = (float*)d_ws;
    float* vbar  = ws;                        // B*D
    float* q1    = vbar + (size_t)B * D;      // B*F
    float* q2    = q1   + (size_t)B * F;      // B*F
    float* scale = q2   + (size_t)B * F;      // F
    float* shift = scale + F;                 // F
    _Float16* Wkh = (_Float16*)(shift + F);   // F*D halves (32B-aligned offset)

    float* outO = (float*)d_out;              // [H,B,D]
    float* outA = outO + (size_t)H * B * D;   // [H,B,S]

    k_cvt_wk <<<(F * D + 255) / 256, 256, 0, stream>>>(Wk, Wkh);
    k_vbar   <<<(B * D + 255) / 256, 256, 0, stream>>>(v, vbar, B);
    k_proj1  <<<B, 128, 0, stream>>>(vbar, Wq, bq, q1);
    k_bnstats<<<F, 256, 0, stream>>>(q1, gamma, beta, scale, shift, B);
    k_proj2  <<<B, 128, 0, stream>>>(q1, scale, shift, W2, b2, q2);
    k_attn   <<<B, 128, 0, stream>>>(v, Wkh, bk, q2, outO, outA, B);
}
